// ResGCN_69947837383264
// MI455X (gfx1250) — compile-verified, matrix operand order checked by reading, not run on hardware
//
#include <hip/hip_runtime.h>
#include <hip/hip_bf16.h>

typedef __attribute__((ext_vector_type(16))) _Float16     v16h;
typedef __attribute__((ext_vector_type(8)))  float        v8f;
typedef __attribute__((ext_vector_type(4)))  unsigned int u32x4;
typedef __attribute__((ext_vector_type(4)))  float        f32x4;

union Frag { v16h v; u32x4 u[2]; };

// ---------------- weight transpose + f16 cast: Wt[n*128+k] = (half)W[k*128+n]
__global__ void k_wt16(const float* __restrict__ W, _Float16* __restrict__ Wt) {
    int i = blockIdx.x * blockDim.x + threadIdx.x;   // exactly 16384 threads
    int k = i >> 7, n = i & 127;
    Wt[(size_t)n * 128 + k] = (_Float16)W[(size_t)k * 128 + n];
}

// ---------------- out-degree (by source), +1 self loop applied in k_dinv
__global__ void k_deg(const int* __restrict__ src, int* __restrict__ deg, int E) {
    int e = blockIdx.x * blockDim.x + threadIdx.x;
    if (e < E) atomicAdd(&deg[src[e]], 1);
}

__global__ void k_dinv(const int* __restrict__ deg, float* __restrict__ dinv, int N) {
    int i = blockIdx.x * blockDim.x + threadIdx.x;
    if (i < N) dinv[i] = rsqrtf((float)deg[i] + 1.0f);
}

// ---------------- column stats (sum, sumsq) with optional relu-on-read
__global__ void k_colstats(const float* __restrict__ h, float* __restrict__ stats,
                           int total, int do_relu) {
    int tid    = blockIdx.x * blockDim.x + threadIdx.x;
    int stride = gridDim.x * blockDim.x;              // multiple of 128
    int col    = tid & 127;
    float s = 0.f, sq = 0.f;
    for (int i = tid; i < total; i += stride) {
        float v = h[i];
        if (do_relu) v = fmaxf(v, 0.f);
        s += v; sq += v * v;
    }
    unsafeAtomicAdd(&stats[col], s);
    unsafeAtomicAdd(&stats[128 + col], sq);
}

__device__ __forceinline__ void bn_coef(const float* stats, const float* g, const float* b,
                                        int col, float invn, float& sc, float& sh) {
    float mean = stats[col] * invn;
    float var  = stats[128 + col] * invn - mean * mean;
    sc = g[col] * rsqrtf(var + 1e-5f);
    sh = b[col] - mean * sc;
}

// ---------------- apply BN (+optional relu-on-read), cast to f16
__global__ void k_bncast(const float* __restrict__ h, const float* __restrict__ stats,
                         const float* __restrict__ g, const float* __restrict__ b,
                         _Float16* __restrict__ out, int total, float invn, int do_relu) {
    int i = blockIdx.x * blockDim.x + threadIdx.x;
    if (i >= total) return;
    int col = i & 127;
    float sc, sh;
    bn_coef(stats, g, b, col, invn, sc, sh);
    float v = h[i];
    if (do_relu) v = fmaxf(v, 0.f);
    out[i] = (_Float16)(v * sc + sh);
}

// ---------------- WMMA GEMM: C[MP x 128] = A16[MP x 128] @ B (Bt col-major 128x128)
// MP is a multiple of 128 (rows padded in workspace; padded rows never read back).
// block = 256 threads = 8 waves; wave w owns the 16-row stripe at blockIdx.x*128 + w*16
// and computes all 8 column tiles, reusing its 4 A k-fragments from registers (32 WMMAs).
// RELU/BIAS are compile-time so the epilogue is branch/select free.
template <int RELU, int BIAS>
__global__ __launch_bounds__(256)
void k_gemm(const _Float16* __restrict__ A, const _Float16* __restrict__ Bt,
            float* __restrict__ C, const float* __restrict__ bias) {
    const int lane = threadIdx.x & 31;
    const int wave = threadIdx.x >> 5;
    const int half = lane >> 4;       // K-group selector per ISA 16-bit layout
    const int lr   = lane & 15;
    const int m0   = blockIdx.x * 128 + wave * 16;

    const _Float16* arow = A + (size_t)(m0 + lr) * 128;
    Frag a[4];
#pragma unroll
    for (int kt = 0; kt < 4; ++kt) {           // preload K=0..127 fragments
        const int kb = kt * 32 + half * 8;     // lanes 0-15: K=kb..kb+7 & kb+16..; lanes 16-31 +8
        a[kt].u[0] = *(const u32x4*)(arow + kb);
        a[kt].u[1] = *(const u32x4*)(arow + kb + 16);
    }

#pragma unroll
    for (int nt = 0; nt < 8; ++nt) {
        const _Float16* brow = Bt + (size_t)(nt * 16 + lr) * 128;
        v8f acc = {};
#pragma unroll
        for (int kt = 0; kt < 4; ++kt) {
            const int kb = kt * 32 + half * 8;
            Frag b;
            b.u[0] = *(const u32x4*)(brow + kb);
            b.u[1] = *(const u32x4*)(brow + kb + 16);
            acc = __builtin_amdgcn_wmma_f32_16x16x32_f16(
                      false, a[kt].v, false, b.v, (short)0, acc, false, false);
        }
        const int col = nt * 16 + lr;
        float ba = 0.f;
        if (BIAS) ba = bias[col];
#pragma unroll
        for (int v = 0; v < 8; ++v) {          // C layout: VGPR v -> row v + half*8
            float val = acc[v];
            if (BIAS) val += ba;
            if (RELU) val = fmaxf(val, 0.f);
            C[(size_t)(m0 + half * 8 + v) * 128 + col] = val;
        }
    }
}

// ---------------- self-loop init + bias: out = xw * dinv^2 + bias
__global__ void k_selfinit(const float* __restrict__ xw, const float* __restrict__ dinv,
                           const float* __restrict__ bias, float* __restrict__ out, int total) {
    int i = blockIdx.x * blockDim.x + threadIdx.x;
    if (i >= total) return;
    int row = i >> 7, col = i & 127;
    float di = dinv[row];
    out[i] = xw[i] * di * di + bias[col];
}

// ---------------- edge scatter: out[dst] += xw[src] * dinv[s]*dinv[d]  (4 feats/thread)
__global__ void k_scatter(const float* __restrict__ xw, const float* __restrict__ dinv,
                          const int* __restrict__ src, const int* __restrict__ dst,
                          float* __restrict__ out, int E) {
    int t = blockIdx.x * blockDim.x + threadIdx.x;
    int e = t >> 5, q = t & 31;
    if (e >= E) return;
    int s = src[e], d = dst[e];
    float w = dinv[s] * dinv[d];
    f32x4 v = *(const f32x4*)(xw + (size_t)s * 128 + q * 4);
    float* o = out + (size_t)d * 128 + q * 4;
    unsafeAtomicAdd(o + 0, v[0] * w);
    unsafeAtomicAdd(o + 1, v[1] * w);
    unsafeAtomicAdd(o + 2, v[2] * w);
    unsafeAtomicAdd(o + 3, v[3] * w);
}

// ---------------- global_add_pool of relu(h) by sorted batch id
__global__ void k_pool(const float* __restrict__ h, const int* __restrict__ batch,
                       float* __restrict__ pooled, int N) {
    int t = blockIdx.x * blockDim.x + threadIdx.x;
    int i = t >> 5, q = t & 31;
    if (i >= N) return;
    int g = batch[i];
    f32x4 v = *(const f32x4*)(h + (size_t)i * 128 + q * 4);
    float* o = pooled + (size_t)g * 128 + q * 4;
    unsafeAtomicAdd(o + 0, fmaxf(v[0], 0.f));
    unsafeAtomicAdd(o + 1, fmaxf(v[1], 0.f));
    unsafeAtomicAdd(o + 2, fmaxf(v[2], 0.f));
    unsafeAtomicAdd(o + 3, fmaxf(v[3], 0.f));
}

// ---------------- final BN + classifier + log_softmax, one block per graph
__global__ __launch_bounds__(128)
void k_cls(const float* __restrict__ fcout, const float* __restrict__ stats,
           const float* __restrict__ g, const float* __restrict__ b,
           const float* __restrict__ Wcls, const float* __restrict__ bcls,
           float* __restrict__ out, float invn, int C) {
    __shared__ float hbn[128];
    __shared__ float logits[32];
    int j = threadIdx.x, gid = blockIdx.x;
    float sc, sh;
    bn_coef(stats, g, b, j, invn, sc, sh);
    hbn[j] = fcout[(size_t)gid * 128 + j] * sc + sh;
    __syncthreads();
    if (j < C) {
        float acc = bcls[j];
        for (int k = 0; k < 128; ++k) acc += hbn[k] * Wcls[k * C + j];
        logits[j] = acc;
    }
    __syncthreads();
    if (j == 0) {
        float m = logits[0];
        for (int c = 1; c < C; ++c) m = fmaxf(m, logits[c]);
        float s = 0.f;
        for (int c = 0; c < C; ++c) s += expf(logits[c] - m);
        float lse = m + logf(s);
        for (int c = 0; c < C; ++c) out[(size_t)gid * C + c] = logits[c] - lse;
    }
}

extern "C" void kernel_launch(void* const* d_in, const int* in_sizes, int n_in,
                              void* d_out, int out_size, void* d_ws, size_t ws_size,
                              hipStream_t stream) {
    // ---- inputs (setup_inputs order)
    const float* x          = (const float*)d_in[0];
    const int*   ei         = (const int*)d_in[1];
    const int*   batch      = (const int*)d_in[2];
    const float* bn_feat_g  = (const float*)d_in[3];
    const float* bn_feat_b  = (const float*)d_in[4];
    const float* W_feat     = (const float*)d_in[5];
    const float* bns_conv_g = (const float*)d_in[6];
    const float* bns_conv_b = (const float*)d_in[7];
    const float* W_conv     = (const float*)d_in[8];
    const float* b_conv     = (const float*)d_in[9];
    const float* bn_fc_g    = (const float*)d_in[10];
    const float* bn_fc_b    = (const float*)d_in[11];
    const float* W_lin      = (const float*)d_in[12];
    const float* b_lin      = (const float*)d_in[13];
    const float* bn_hid_g   = (const float*)d_in[14];
    const float* bn_hid_b   = (const float*)d_in[15];
    const float* W_cls      = (const float*)d_in[16];
    const float* b_cls      = (const float*)d_in[17];

    const int N  = in_sizes[2];
    const int E  = in_sizes[1] / 2;
    const int C  = in_sizes[17];
    const int G  = out_size / C;
    const int L  = in_sizes[6] / 128;
    const int NP = ((N + 127) / 128) * 128;   // padded GEMM rows (no guards needed)
    const int GP = ((G + 127) / 128) * 128;
    const int* src = ei;
    const int* dst = ei + E;

    // ---- workspace carve-up (256B aligned); GEMM operands padded to NP/GP rows
    char* base = (char*)d_ws;
    size_t off = 0;
    auto bump = [&](size_t bytes) { size_t o = off; off = (off + bytes + 255) & ~(size_t)255; return o; };
    float*    h        = (float*)   (base + bump((size_t)NP * 128 * 4));
    float*    xw       = (float*)   (base + bump((size_t)NP * 128 * 4));
    _Float16* h16      = (_Float16*)(base + bump((size_t)NP * 128 * 2));
    _Float16* wt       = (_Float16*)(base + bump((size_t)(2 + L) * 128 * 128 * 2)); // feat, conv0..L-1, lin
    int*      deg      = (int*)     (base + bump((size_t)N * 4));
    float*    dinv     = (float*)   (base + bump((size_t)N * 4));
    float*    stats    = (float*)   (base + bump((size_t)(L + 3) * 256 * 4));
    float*    pooled   = (float*)   (base + bump((size_t)G * 128 * 4));
    _Float16* pooled16 = (_Float16*)(base + bump((size_t)GP * 128 * 2));
    float*    fcout    = (float*)   (base + bump((size_t)GP * 128 * 4));
    (void)n_in; (void)ws_size;

    _Float16* wt_feat = wt;
    _Float16* wt_lin  = wt + (size_t)(1 + L) * 16384;

    // ---- zero accumulators (re-done every replay)
    hipMemsetAsync(deg,    0, (size_t)N * 4, stream);
    hipMemsetAsync(stats,  0, (size_t)(L + 3) * 256 * 4, stream);
    hipMemsetAsync(pooled, 0, (size_t)G * 128 * 4, stream);

    // ---- weights -> f16, transposed to column-major for contiguous-K B loads
    k_wt16<<<64, 256, 0, stream>>>(W_feat, wt_feat);
    for (int l = 0; l < L; ++l)
        k_wt16<<<64, 256, 0, stream>>>(W_conv + (size_t)l * 16384, wt + (size_t)(1 + l) * 16384);
    k_wt16<<<64, 256, 0, stream>>>(W_lin, wt_lin);

    // ---- symmetric normalization coefficients
    k_deg<<<(E + 255) / 256, 256, 0, stream>>>(src, deg, E);
    k_dinv<<<(N + 255) / 256, 256, 0, stream>>>(deg, dinv, N);

    const int   totN = N * 128;
    const float invN = 1.0f / (float)N;

    // ---- feature BN + feat GEMM (relu fused)
    k_colstats<<<512, 256, 0, stream>>>(x, stats, totN, 0);
    k_bncast<<<(totN + 255) / 256, 256, 0, stream>>>(x, stats, bn_feat_g, bn_feat_b, h16, totN, invN, 0);
    k_gemm<1, 0><<<NP / 128, 256, 0, stream>>>(h16, wt_feat, h, nullptr);

    // ---- GCN layers: BN -> GEMM -> self-loop+bias -> edge scatter (relu deferred to readers)
    for (int l = 0; l < L; ++l) {
        float* sl = stats + (size_t)256 * (1 + l);
        int dorelu = (l > 0);   // h from layer>0 is raw conv output; layer 0 input already relu'd
        k_colstats<<<512, 256, 0, stream>>>(h, sl, totN, dorelu);
        k_bncast<<<(totN + 255) / 256, 256, 0, stream>>>(
            h, sl, bns_conv_g + (size_t)l * 128, bns_conv_b + (size_t)l * 128, h16, totN, invN, dorelu);
        k_gemm<0, 0><<<NP / 128, 256, 0, stream>>>(h16, wt + (size_t)(1 + l) * 16384, xw, nullptr);
        k_selfinit<<<(totN + 255) / 256, 256, 0, stream>>>(xw, dinv, b_conv + (size_t)l * 128, h, totN);
        k_scatter<<<(E * 32 + 255) / 256, 256, 0, stream>>>(xw, dinv, src, dst, h, E);
    }

    // ---- pool relu(h) per graph
    k_pool<<<(N * 32 + 255) / 256, 256, 0, stream>>>(h, batch, pooled, N);

    // ---- FC head
    float* s_fc  = stats + (size_t)256 * (1 + L);
    float* s_hid = stats + (size_t)256 * (2 + L);
    const int   totG = G * 128;
    const float invG = 1.0f / (float)G;
    int gridG = (totG + 255) / 256;

    k_colstats<<<gridG, 256, 0, stream>>>(pooled, s_fc, totG, 0);
    k_bncast<<<gridG, 256, 0, stream>>>(pooled, s_fc, bn_fc_g, bn_fc_b, pooled16, totG, invG, 0);
    k_gemm<1, 1><<<GP / 128, 256, 0, stream>>>(pooled16, wt_lin, fcout, b_lin);
    k_colstats<<<gridG, 256, 0, stream>>>(fcout, s_hid, totG, 0);

    // ---- hidden BN + classifier + log_softmax
    k_cls<<<G, 128, 0, stream>>>(fcout, s_hid, bn_hid_g, bn_hid_b, W_cls, b_cls,
                                 (float*)d_out, invG, C);
}
